// SANLayer_29257317220552
// MI455X (gfx1250) — matrix operand order repaired and checked by reading, not compiled, not run on hardware
//
#include <hip/hip_runtime.h>

#define D_DIMS 2
#define NN     4096
#define FF     128

typedef __attribute__((ext_vector_type(16))) __bf16 v16bf;
typedef __attribute__((ext_vector_type(8)))  __bf16 v8bf;
typedef __attribute__((ext_vector_type(8)))  float  v8f;

// ---------- helpers ----------------------------------------------------------

__device__ __forceinline__ __bf16 f2bf(float f) {
    return (__bf16)f;                               // native v_cvt_*_bf16_f32
}
__device__ __forceinline__ unsigned short f2bf_u(float f) {
    return __builtin_bit_cast(unsigned short, (__bf16)f);
}
__device__ __forceinline__ unsigned pk2(float lo, float hi) {
    return (unsigned)f2bf_u(lo) | ((unsigned)f2bf_u(hi) << 16);
}
__device__ __forceinline__ v16bf mk16(uint4 a, uint4 b) {
    v8bf lo = __builtin_bit_cast(v8bf, a);
    v8bf hi = __builtin_bit_cast(v8bf, b);
    return __builtin_shufflevector(lo, hi, 0,1,2,3,4,5,6,7,8,9,10,11,12,13,14,15);
}
__device__ __forceinline__ v8f wmma_bf16(v16bf a, v16bf b, v8f c) {
    return __builtin_amdgcn_wmma_f32_16x16x32_bf16(false, a, false, b,
                                                   (short)0, c, false, false);
}
// flat shared pointer -> LDS byte offset (aperture: addr[31:0] == LDS_ADDR)
__device__ __forceinline__ unsigned lds_off(void* p) {
    return (unsigned)(unsigned long long)p;
}
// per-lane async DMA: 16B global -> LDS, tracked by ASYNCcnt
__device__ __forceinline__ void async_b128(unsigned lds, const void* g) {
    asm volatile("global_load_async_to_lds_b128 %0, %1, off"
                 :: "v"(lds), "v"(g) : "memory");
}

// ---------- kernel 0: h = X @ W for {Wp, Ws, Wu, Wd} -------------------------
// block = one 16-row strip of X, 8 waves cover the 8 col tiles. W^T staged
// once in LDS as bf16 (row stride 272B => conflict-free b128 fragment reads).

__global__ void __launch_bounds__(256) xw_gemm(
        const float* __restrict__ X,
        const float* __restrict__ Wp, const float* __restrict__ Ws,
        const float* __restrict__ Wu, const float* __restrict__ Wd,
        float* __restrict__ out,
        unsigned short* __restrict__ xwpT,
        float* __restrict__ h_u, float* __restrict__ h_d) {
    __shared__ unsigned short ldsW[128][136];     // W^T: [n][k], padded

    int blk    = blockIdx.x;
    int rowblk = blk & 255;
    int mid    = (blk >> 8) & 3;
    int d      = blk >> 10;

    const float* W = (mid == 0 ? Wp : mid == 1 ? Ws : mid == 2 ? Wu : Wd)
                     + (size_t)d * FF * FF;
    const float* x = X + (size_t)d * NN * FF;

    // stage W^T (bf16) into LDS: thread t loads half a row of W, coalesced
    {
        int r = threadIdx.x >> 1;
        int cbase = (threadIdx.x & 1) * 64;
        const float* wr = W + (size_t)r * FF + cbase;
#pragma unroll
        for (int c = 0; c < 64; c += 4) {
            float4 wv = *reinterpret_cast<const float4*>(wr + c);
            ldsW[cbase + c + 0][r] = f2bf_u(wv.x);
            ldsW[cbase + c + 1][r] = f2bf_u(wv.y);
            ldsW[cbase + c + 2][r] = f2bf_u(wv.z);
            ldsW[cbase + c + 3][r] = f2bf_u(wv.w);
        }
    }
    __syncthreads();

    int wave = threadIdx.x >> 5, lane = threadIdx.x & 31;
    int sel = lane >> 4, ln = lane & 15;
    int m0 = rowblk * 16;
    int ng = wave * 16 + ln;                      // this lane's output column

    v8f acc = {0.f,0.f,0.f,0.f,0.f,0.f,0.f,0.f};

#pragma unroll
    for (int kk = 0; kk < 4; ++kk) {
        int k0 = kk * 32;
        // A fragment: row m0+ln, K = {sel*8..+7} u {16+sel*8..+7}
        const float* ar = x + (size_t)(m0 + ln) * FF + k0 + sel * 8;
        float4 p0 = *reinterpret_cast<const float4*>(ar);
        float4 p1 = *reinterpret_cast<const float4*>(ar + 4);
        float4 p2 = *reinterpret_cast<const float4*>(ar + 16);
        float4 p3 = *reinterpret_cast<const float4*>(ar + 20);
        v16bf a;
        a[0]=f2bf(p0.x); a[1]=f2bf(p0.y); a[2]=f2bf(p0.z); a[3]=f2bf(p0.w);
        a[4]=f2bf(p1.x); a[5]=f2bf(p1.y); a[6]=f2bf(p1.z); a[7]=f2bf(p1.w);
        a[8]=f2bf(p2.x); a[9]=f2bf(p2.y); a[10]=f2bf(p2.z); a[11]=f2bf(p2.w);
        a[12]=f2bf(p3.x); a[13]=f2bf(p3.y); a[14]=f2bf(p3.z); a[15]=f2bf(p3.w);
        // B fragment: W^T row ng, K = {sel*8..+7} u {16+sel*8..+7}  (2 x ds_load_b128)
        uint4 q0 = *reinterpret_cast<const uint4*>(&ldsW[ng][k0 + sel * 8]);
        uint4 q1 = *reinterpret_cast<const uint4*>(&ldsW[ng][k0 + 16 + sel * 8]);
        acc = wmma_bf16(a, mk16(q0, q1), acc);
    }

    if (mid == 0) {
        // store transposed bf16: XWpT[d][n][m], 8 contiguous m per lane
        uint4 o;
        o.x = pk2(acc[0], acc[1]);
        o.y = pk2(acc[2], acc[3]);
        o.z = pk2(acc[4], acc[5]);
        o.w = pk2(acc[6], acc[7]);
        *reinterpret_cast<uint4*>(
            &xwpT[((size_t)d * FF + ng) * NN + m0 + 8 * sel]) = o;
    } else {
#pragma unroll
        for (int v = 0; v < 8; ++v) {
            int m = m0 + 8 * sel + v;                 // acc[v] holds M=v+8*sel
            size_t idx = ((size_t)d * NN + m) * FF + ng;
            if (mid == 1)      out[idx] = tanhf(acc[v]);
            else if (mid == 2) h_u[idx] = acc[v];
            else               h_d[idx] = acc[v];
        }
    }
}

// ---------- kernel 1: s1/s2 = h @ a1, h @ a2 ---------------------------------

__global__ void s1s2_kernel(const float* __restrict__ h_u, const float* __restrict__ h_d,
                            const float* __restrict__ au1, const float* __restrict__ au2,
                            const float* __restrict__ ad1, const float* __restrict__ ad2,
                            float* __restrict__ s1u, float* __restrict__ s2u,
                            float* __restrict__ s1d, float* __restrict__ s2d) {
    int wave = threadIdx.x >> 5, lane = threadIdx.x & 31;
    int gw = blockIdx.x * 8 + wave;          // 0..16383
    int d = gw >> 13;
    int g = (gw >> 12) & 1;
    int i = gw & 4095;

    const float* h  = (g ? h_d : h_u) + ((size_t)d * NN + i) * FF;
    const float* a1 = (g ? ad1 : au1) + (size_t)d * FF;
    const float* a2 = (g ? ad2 : au2) + (size_t)d * FF;

    int f0 = lane * 4;
    float4 hv = *reinterpret_cast<const float4*>(h + f0);
    float4 A1 = *reinterpret_cast<const float4*>(a1 + f0);
    float4 A2 = *reinterpret_cast<const float4*>(a2 + f0);
    float p1 = hv.x*A1.x + hv.y*A1.y + hv.z*A1.z + hv.w*A1.w;
    float p2 = hv.x*A2.x + hv.y*A2.y + hv.z*A2.z + hv.w*A2.w;
#pragma unroll
    for (int off = 16; off > 0; off >>= 1) {
        p1 += __shfl_xor(p1, off, 32);
        p2 += __shfl_xor(p2, off, 32);
    }
    if (lane == 0) {
        (g ? s1d : s1u)[(size_t)d * NN + i] = p1;
        (g ? s2d : s2u)[(size_t)d * NN + i] = p2;
    }
}

// ---------- kernel 2: out += tanh(L @ XWp) -----------------------------------
// block tile 32(M) x 128(N), K=4096. L (f32) staged through regs -> bf16 LDS;
// B tile (bf16, L2-hot) double-buffered in LDS via ASYNC global->LDS DMA so
// the synchronous VMEM pipe is dedicated to the 134MB L stream.

__global__ void __launch_bounds__(256) hp_gemm(const float* __restrict__ L,
                                               const unsigned short* __restrict__ xwpT,
                                               float* __restrict__ out) {
    __shared__ unsigned short ldsA[32][40];        // 80B stride: conflict-free
    __shared__ unsigned short ldsB[2][128][40];    // double-buffered B tile

    int d    = blockIdx.x >> 7;
    int mblk = blockIdx.x & 127;
    int m0   = mblk * 32;
    const float* Lrow = L + ((size_t)d * NN + m0) * NN;

    int wave = threadIdx.x >> 5, lane = threadIdx.x & 31;
    int sel = lane >> 4, ln = lane & 15;
    int mtile = wave & 1;
    int npair = wave >> 1;                    // 0..3
    int n0g = npair * 32 + ln;
    int n1g = n0g + 16;

    int lr = threadIdx.x >> 3;                // L staging: row 0..31
    int lc = (threadIdx.x & 7) * 4;           // col 0..28

    // async B staging coords: thread t copies 32B of row tn
    int tn  = threadIdx.x >> 1;               // 0..127
    int th2 = (threadIdx.x & 1) * 16;         // halves offset (0 or 16)
    const unsigned short* bsrc = xwpT + ((size_t)d * FF + tn) * NN + th2;

    v8f acc0 = {0.f,0.f,0.f,0.f,0.f,0.f,0.f,0.f};
    v8f acc1 = {0.f,0.f,0.f,0.f,0.f,0.f,0.f,0.f};

    // prologue: async-fill buffer 0 with B(k=0)
    async_b128(lds_off(&ldsB[0][tn][th2]),     bsrc);
    async_b128(lds_off(&ldsB[0][tn][th2 + 8]), bsrc + 8);

    for (int k0 = 0; k0 < NN; k0 += 32) {
        int buf = (k0 >> 5) & 1;
        // L tile for this step -> registers (global pipe)
        float4 lv = *reinterpret_cast<const float4*>(
            Lrow + (size_t)lr * NN + k0 + lc);
        if (k0 + 256 < NN)
            __builtin_prefetch(Lrow + (size_t)lr * NN + k0 + 256 + lc, 0, 0);

        __syncthreads();   // all waves done with prev compute: A & buf^1 free
        bool more = (k0 + 32 < NN);
        if (more) {        // kick off async DMA of the next B tile
            async_b128(lds_off(&ldsB[buf ^ 1][tn][th2]),     bsrc + k0 + 32);
            async_b128(lds_off(&ldsB[buf ^ 1][tn][th2 + 8]), bsrc + k0 + 40);
        }
        // stage A tile as bf16
        uint2 pk;
        pk.x = pk2(lv.x, lv.y);
        pk.y = pk2(lv.z, lv.w);
        *reinterpret_cast<uint2*>(&ldsA[lr][lc]) = pk;
        // current B tile's 2 copies complete (in-order; next tile's 2 pending)
        if (more) asm volatile("s_wait_asynccnt 2" ::: "memory");
        else      asm volatile("s_wait_asynccnt 0" ::: "memory");
        __syncthreads();

        // A fragment from LDS: K = sel*16..+15 (contiguous order, same as B)
        int ml = mtile * 16 + ln;
        uint4 q0 = *reinterpret_cast<const uint4*>(&ldsA[ml][sel * 16]);
        uint4 q1 = *reinterpret_cast<const uint4*>(&ldsA[ml][sel * 16 + 8]);
        v16bf a = mk16(q0, q1);

        // B fragments from the ready LDS buffer
        uint4 r0 = *reinterpret_cast<const uint4*>(&ldsB[buf][n0g][sel * 16]);
        uint4 r1 = *reinterpret_cast<const uint4*>(&ldsB[buf][n0g][sel * 16 + 8]);
        uint4 r2 = *reinterpret_cast<const uint4*>(&ldsB[buf][n1g][sel * 16]);
        uint4 r3 = *reinterpret_cast<const uint4*>(&ldsB[buf][n1g][sel * 16 + 8]);
        acc0 = wmma_bf16(a, mk16(r0, r1), acc0);
        acc1 = wmma_bf16(a, mk16(r2, r3), acc1);
    }

#pragma unroll
    for (int v = 0; v < 8; ++v) {
        int m = m0 + mtile * 16 + 8 * sel + v;
        float* o = out + ((size_t)d * NN + m) * FF;
        o[n0g] += tanhf(acc0[v]);
        o[n1g] += tanhf(acc1[v]);
    }
}

// ---------- kernel 3: out += tanh(sparse-softmax(adj, scores) @ h) -----------

#define CAP 192

__global__ void gat_aggregate(const float* __restrict__ Lu, const float* __restrict__ Ld,
                              const float* __restrict__ h_u, const float* __restrict__ h_d,
                              const float* __restrict__ s1u, const float* __restrict__ s2u,
                              const float* __restrict__ s1d, const float* __restrict__ s2d,
                              float* __restrict__ out) {
    __shared__ int   s_j[8][CAP];
    __shared__ float s_e[8][CAP];

    int wave = threadIdx.x >> 5, lane = threadIdx.x & 31;
    int gw = blockIdx.x * 8 + wave;          // 0..16383
    int d = gw >> 13;
    int g = (gw >> 12) & 1;
    int i = gw & 4095;

    const float* adj  = (g ? Ld : Lu) + ((size_t)d * NN + i) * NN;
    const float* hmat = (g ? h_d : h_u) + (size_t)d * NN * FF;
    const float* s2   = (g ? s2d : s2u) + (size_t)d * NN;
    float s1i = (g ? s1d : s1u)[(size_t)d * NN + i];

    int cnt = 0;
    float mx = -3.0e38f;
    for (int j0 = 0; j0 < NN; j0 += 32) {
        int j = j0 + lane;
        float mval = adj[j];
        bool hit = (mval == 1.0f);
        float e = 0.f;
        if (hit) {
            float s = s1i + s2[j];
            e = (s > 0.f) ? s : 0.01f * s;   // LeakyReLU(0.01)
            mx = fmaxf(mx, e);
        }
        unsigned bal = __builtin_amdgcn_ballot_w32(hit);
        if (hit) {
            int pos = cnt + __popc(bal & ((1u << lane) - 1u));
            if (pos < CAP) { s_j[wave][pos] = j; s_e[wave][pos] = e; }
        }
        cnt += __popc(bal);
    }
    if (cnt > CAP) cnt = CAP;
    if (cnt == 0) return;                    // empty row contributes tanh(0)=0

#pragma unroll
    for (int off = 16; off > 0; off >>= 1) mx = fmaxf(mx, __shfl_xor(mx, off, 32));

    float sum = 0.f;
    for (int p = lane; p < cnt; p += 32) {
        float w = __expf(s_e[wave][p] - mx);
        s_e[wave][p] = w;
        sum += w;
    }
#pragma unroll
    for (int off = 16; off > 0; off >>= 1) sum += __shfl_xor(sum, off, 32);
    float inv = 1.0f / sum;

    int f0 = lane * 4;
    float a0 = 0.f, a1 = 0.f, a2 = 0.f, a3 = 0.f;
    for (int p = 0; p < cnt; ++p) {
        int j = s_j[wave][p];
        float w = s_e[wave][p] * inv;
        float4 hv = *reinterpret_cast<const float4*>(&hmat[(size_t)j * FF + f0]);
        a0 += w * hv.x; a1 += w * hv.y; a2 += w * hv.z; a3 += w * hv.w;
    }
    float4* o = reinterpret_cast<float4*>(&out[((size_t)d * NN + i) * FF + f0]);
    float4 cur = *o;
    cur.x += tanhf(a0); cur.y += tanhf(a1); cur.z += tanhf(a2); cur.w += tanhf(a3);
    *o = cur;
}

// ---------- launch -----------------------------------------------------------

extern "C" void kernel_launch(void* const* d_in, const int* in_sizes, int n_in,
                              void* d_out, int out_size, void* d_ws, size_t ws_size,
                              hipStream_t stream) {
    const float* X   = (const float*)d_in[0];
    const float* L   = (const float*)d_in[1];
    const float* Lu  = (const float*)d_in[2];
    const float* Ld  = (const float*)d_in[3];
    const float* Wp  = (const float*)d_in[4];
    const float* Ws  = (const float*)d_in[5];
    const float* Wu  = (const float*)d_in[6];
    const float* au1 = (const float*)d_in[7];
    const float* au2 = (const float*)d_in[8];
    const float* Wd  = (const float*)d_in[9];
    const float* ad1 = (const float*)d_in[10];
    const float* ad2 = (const float*)d_in[11];
    float* out = (float*)d_out;

    char* ws = (char*)d_ws;
    unsigned short* xwpT = (unsigned short*)ws;                // 2 MB
    float* h_u = (float*)(ws + (2u << 20));                    // 4 MB
    float* h_d = (float*)(ws + (6u << 20));                    // 4 MB
    float* s1u = (float*)(ws + (10u << 20));
    float* s2u = s1u + D_DIMS * NN;
    float* s1d = s2u + D_DIMS * NN;
    float* s2d = s1d + D_DIMS * NN;

    xw_gemm<<<dim3(D_DIMS * 4 * 256), dim3(256), 0, stream>>>(
        X, Wp, Ws, Wu, Wd, out, xwpT, h_u, h_d);
    s1s2_kernel<<<dim3(2048), dim3(256), 0, stream>>>(
        h_u, h_d, au1, au2, ad1, ad2, s1u, s2u, s1d, s2d);
    hp_gemm<<<dim3(D_DIMS * 128), dim3(256), 0, stream>>>(L, xwpT, out);
    gat_aggregate<<<dim3(2048), dim3(256), 0, stream>>>(
        Lu, Ld, h_u, h_d, s1u, s2u, s1d, s2d, out);
}
